// StandardAttention_37495064494482
// MI455X (gfx1250) — compile-verified, matrix operand order checked by reading, not run on hardware
//
#include <hip/hip_runtime.h>

typedef __attribute__((ext_vector_type(16))) _Float16 v16h;
typedef __attribute__((ext_vector_type(8)))  _Float16 v8h;
typedef __attribute__((ext_vector_type(4)))  _Float16 v4h;
typedef __attribute__((ext_vector_type(8)))  float    v8f;

#define N_HEAD   16
#define HEAD_DIM 64
#define N_EMBD   1024
#define BATCH    2
#define SEQ      2048

#define WMMA_F16(A, B, C) \
  __builtin_amdgcn_wmma_f32_16x16x32_f16(false, (A), false, (B), (short)0, (C), false, false)

// Load a 16x32 f16 fragment whose rows are contiguous in memory (A-matrix
// style; also valid for B when B[k][n] = W[n][k], i.e. lane n reads row n of W).
// ISA 7.12.2 layout: lanes 0-15 row=lane, K in {0..7, 16..23};
//                    lanes 16-31 row=lane-16, K in {8..15, 24..31}.
__device__ __forceinline__ v16h load_frag_rows(const _Float16* __restrict__ base,
                                               int ld, int lane) {
  const int r  = lane & 15;
  const int kh = (lane >> 4) & 1;
  const _Float16* p = base + (size_t)r * ld + kh * 8;
  v8h lo = *(const v8h*)(p);
  v8h hi = *(const v8h*)(p + 16);
  v16h f;
#pragma unroll
  for (int i = 0; i < 8; ++i) { f[i] = lo[i]; f[i + 8] = hi[i]; }
  return f;
}

// ---------------- fp32 -> f16 conversion (vectorized x4) ----------------
__global__ void __launch_bounds__(256)
cvt_f32_to_f16_x4(const float* __restrict__ src, _Float16* __restrict__ dst, int n4) {
  int i = blockIdx.x * blockDim.x + threadIdx.x;
  if (i >= n4) return;
  float4 v = ((const float4*)src)[i];
  v4h o;
  o[0] = (_Float16)v.x; o[1] = (_Float16)v.y;
  o[2] = (_Float16)v.z; o[3] = (_Float16)v.w;
  ((v4h*)dst)[i] = o;
}

// ---------------- shared 32x64 wave GEMM core (K = N_EMBD, y = A @ W^T) ------
// Explicit ping-pong software pipeline (k-loop unrolled x2): each fragment set
// is loaded straight into its own registers and consumed in place -> no
// rotation moves, and the loadcnt wait before a WMMA covers loads issued a
// full 32-wide k-step earlier.
struct GFrags { v16h a0, a1, b0, b1, b2, b3; };

__device__ __forceinline__ GFrags load_gfrags(const _Float16* __restrict__ A,
                                              const _Float16* __restrict__ B,
                                              int k, int lane) {
  GFrags f;
  f.a0 = load_frag_rows(A + k,               N_EMBD, lane);
  f.a1 = load_frag_rows(A + 16 * N_EMBD + k, N_EMBD, lane);
  f.b0 = load_frag_rows(B + k,               N_EMBD, lane);
  f.b1 = load_frag_rows(B + 16 * N_EMBD + k, N_EMBD, lane);
  f.b2 = load_frag_rows(B + 32 * N_EMBD + k, N_EMBD, lane);
  f.b3 = load_frag_rows(B + 48 * N_EMBD + k, N_EMBD, lane);
  return f;
}

__device__ __forceinline__ void mma_gfrags(const GFrags& f, v8f acc[2][4]) {
  acc[0][0] = WMMA_F16(f.a0, f.b0, acc[0][0]);
  acc[0][1] = WMMA_F16(f.a0, f.b1, acc[0][1]);
  acc[0][2] = WMMA_F16(f.a0, f.b2, acc[0][2]);
  acc[0][3] = WMMA_F16(f.a0, f.b3, acc[0][3]);
  acc[1][0] = WMMA_F16(f.a1, f.b0, acc[1][0]);
  acc[1][1] = WMMA_F16(f.a1, f.b1, acc[1][1]);
  acc[1][2] = WMMA_F16(f.a1, f.b2, acc[1][2]);
  acc[1][3] = WMMA_F16(f.a1, f.b3, acc[1][3]);
}

__device__ __forceinline__ void wave_gemm_32x64(const _Float16* __restrict__ Abase,
                                                const _Float16* __restrict__ Bbase,
                                                int lane, v8f acc[2][4]) {
  GFrags ping = load_gfrags(Abase, Bbase, 0, lane);
#pragma unroll 1
  for (int k0 = 0; k0 < N_EMBD - 64; k0 += 64) {
    GFrags pong = load_gfrags(Abase, Bbase, k0 + 32, lane);
    mma_gfrags(ping, acc);                       // consumes k0
    ping = load_gfrags(Abase, Bbase, k0 + 64, lane);
    mma_gfrags(pong, acc);                       // consumes k0 + 32
  }
  GFrags pong = load_gfrags(Abase, Bbase, N_EMBD - 32, lane);
  mma_gfrags(ping, acc);                         // consumes N_EMBD - 64
  mma_gfrags(pong, acc);                         // consumes N_EMBD - 32
}

// ---------------- QKV projection: qkv = x @ w_attn^T ----------------
// M = B*T = 4096, N = 3C = 3072, K = C = 1024.
// Wave computes a 32x64 tile. Q,K stored [B,H,T,D]; V stored transposed [B,H,D,T].
__global__ void __launch_bounds__(128)
qkv_gemm(const _Float16* __restrict__ xb, const _Float16* __restrict__ wab,
         _Float16* __restrict__ qb, _Float16* __restrict__ kb,
         _Float16* __restrict__ vtb) {
  const int lane  = threadIdx.x & 31;
  const int wv    = threadIdx.x >> 5;                 // 0..3
  const int mbase = blockIdx.x * 32;                  // M base (0..4064)
  const int nbase = blockIdx.y * 256 + wv * 64;       // N base for this wave

  v8f acc[2][4] = {};
  wave_gemm_32x64(xb + (size_t)mbase * N_EMBD,
                  wab + (size_t)nbase * N_EMBD, lane, acc);

  // C/D layout: lane holds column n = nbase + t*16 + (lane&15),
  // reg r holds row m = mbase + i*16 + r + 8*(lane>>4).
  const int col = lane & 15;
  const int rb  = (lane >> 4) * 8;
#pragma unroll
  for (int i = 0; i < 2; ++i) {
#pragma unroll
    for (int t = 0; t < 4; ++t) {
#pragma unroll
      for (int r = 0; r < 8; ++r) {
        int m  = mbase + i * 16 + rb + r;
        int n  = nbase + t * 16 + col;
        int b_ = m >> 11;                // batch
        int tt = m & (SEQ - 1);          // time
        int wh = n >> 10;                // 0=q 1=k 2=v
        int c  = n & (N_EMBD - 1);
        int h  = c >> 6, d = c & (HEAD_DIM - 1);
        _Float16 val = (_Float16)acc[i][t][r];
        size_t bh = (size_t)(b_ * N_HEAD + h);
        if (wh == 0)      qb [(bh * SEQ + tt) * HEAD_DIM + d] = val;
        else if (wh == 1) kb [(bh * SEQ + tt) * HEAD_DIM + d] = val;
        else              vtb[(bh * HEAD_DIM + d) * SEQ + tt] = val;
      }
    }
  }
}

// ---------------- Flash attention (transposed-score formulation) ----------------
// Per wave: 16 queries of one (b,h). Computes S^T = K Q^T so the softmax over
// keys is per-lane (+one shfl_xor(16)), and P lands directly in the B-fragment
// layout for y^T = V^T P (A-frags from pre-transposed V, contiguous rows).
struct KFrags { v16h k00, k01, k10, k11; };

__device__ __forceinline__ KFrags load_kfrags(const _Float16* __restrict__ kbh,
                                              int kb0, int lane) {
  const _Float16* krow0 = kbh + (size_t)(kb0)      * HEAD_DIM;
  const _Float16* krow1 = kbh + (size_t)(kb0 + 16) * HEAD_DIM;
  KFrags f;
  f.k00 = load_frag_rows(krow0 + 0,  HEAD_DIM, lane);
  f.k01 = load_frag_rows(krow0 + 32, HEAD_DIM, lane);
  f.k10 = load_frag_rows(krow1 + 0,  HEAD_DIM, lane);
  f.k11 = load_frag_rows(krow1 + 32, HEAD_DIM, lane);
  return f;
}

// One 32-key flash step: S^T WMMAs, V fetch (overlaps softmax VALU),
// online-softmax update, accumulator rescale + WMMAs.
__device__ __forceinline__ void attn_step(const KFrags& kf, int kb0,
                                          const v16h& bq0, const v16h& bq1,
                                          const _Float16* __restrict__ vbh,
                                          int lane, v8f acc[4],
                                          float& m_run, float& l_run) {
  const float scale = 0.125f;      // 1/sqrt(64)
  v8f s0 = {}, s1 = {};
  s0 = WMMA_F16(kf.k00, bq0, s0);
  s0 = WMMA_F16(kf.k01, bq1, s0);
  s1 = WMMA_F16(kf.k10, bq0, s1);
  s1 = WMMA_F16(kf.k11, bq1, s1);

  // V^T fragments for this key step — independent of the softmax below,
  // so their latency hides under the exp/pack VALU work.
  v16h va[4];
#pragma unroll
  for (int t = 0; t < 4; ++t)
    va[t] = load_frag_rows(vbh + (size_t)(t * 16) * SEQ + kb0, SEQ, lane);

  // Online softmax: lane holds 16 key-scores of query (lane&15); the other
  // 16 keys for the same query live in lane^16.
  float tmax = -3.0e38f;
#pragma unroll
  for (int i = 0; i < 8; ++i) {
    s0[i] *= scale; s1[i] *= scale;
    tmax = fmaxf(tmax, fmaxf(s0[i], s1[i]));
  }
  tmax = fmaxf(tmax, __shfl_xor(tmax, 16));
  float m_new = fmaxf(m_run, tmax);
  float alpha = __expf(m_run - m_new);

  float rsum = 0.0f;
  v16h pf;                          // P in B-fragment layout (see ISA 7.12.2)
#pragma unroll
  for (int i = 0; i < 8; ++i) {
    float p0 = __expf(s0[i] - m_new);
    float p1 = __expf(s1[i] - m_new);
    rsum += p0 + p1;
    pf[i]     = (_Float16)p0;       // keys {0..7} / {8..15}  of this step
    pf[i + 8] = (_Float16)p1;       // keys {16..23} / {24..31}
  }
  rsum += __shfl_xor(rsum, 16);
  l_run = l_run * alpha + rsum;
  m_run = m_new;

  // acc(d,q) = alpha*acc + V^T(d, keys) x P(keys, q)
#pragma unroll
  for (int t = 0; t < 4; ++t) {
#pragma unroll
    for (int i = 0; i < 8; ++i) acc[t][i] *= alpha;
    acc[t] = WMMA_F16(va[t], pf, acc[t]);
  }
}

__global__ void __launch_bounds__(128)
attn_flash(const _Float16* __restrict__ qb, const _Float16* __restrict__ kb,
           const _Float16* __restrict__ vtb, _Float16* __restrict__ yb) {
  const int lane = threadIdx.x & 31;
  const int wv   = threadIdx.x >> 5;
  const int bh   = blockIdx.x;                 // 0..31
  const int qt   = blockIdx.y * 4 + wv;        // query tile 0..127

  const _Float16* qbh = qb  + (size_t)bh * SEQ * HEAD_DIM;
  const _Float16* kbh = kb  + (size_t)bh * SEQ * HEAD_DIM;
  const _Float16* vbh = vtb + (size_t)bh * HEAD_DIM * SEQ;

  // Loop-invariant Q fragments (d = 0..31 and 32..63)
  const _Float16* qrow = qbh + (size_t)(qt * 16) * HEAD_DIM;
  const v16h bq0 = load_frag_rows(qrow + 0,  HEAD_DIM, lane);
  const v16h bq1 = load_frag_rows(qrow + 32, HEAD_DIM, lane);

  v8f acc[4] = {};                 // y^T tiles: rows d (4x16), col = query
  float m_run = -3.0e38f;
  float l_run = 0.0f;

  // Ping-pong over 32-key steps: next step's K fragments are fetched while the
  // current step's softmax + accumulate WMMAs execute.
  KFrags ping = load_kfrags(kbh, 0, lane);
#pragma unroll 1
  for (int kb0 = 0; kb0 < SEQ - 64; kb0 += 64) {
    KFrags pong = load_kfrags(kbh, kb0 + 32, lane);
    attn_step(ping, kb0, bq0, bq1, vbh, lane, acc, m_run, l_run);
    ping = load_kfrags(kbh, kb0 + 64, lane);
    attn_step(pong, kb0 + 32, bq0, bq1, vbh, lane, acc, m_run, l_run);
  }
  KFrags pong = load_kfrags(kbh, SEQ - 32, lane);
  attn_step(ping, SEQ - 64, bq0, bq1, vbh, lane, acc, m_run, l_run);
  attn_step(pong, SEQ - 32, bq0, bq1, vbh, lane, acc, m_run, l_run);

  // Normalize and store y as [B,T,C] f16 for the projection GEMM.
  const float inv = 1.0f / l_run;
  const int q  = lane & 15;
  const int rb = (lane >> 4) * 8;
  const int b_ = bh >> 4, h = bh & 15;
  const size_t mrow = (size_t)b_ * SEQ + qt * 16 + q;
#pragma unroll
  for (int t = 0; t < 4; ++t) {
#pragma unroll
    for (int r = 0; r < 8; ++r) {
      int d = t * 16 + rb + r;
      yb[mrow * N_EMBD + h * HEAD_DIM + d] = (_Float16)(acc[t][r] * inv);
    }
  }
}

// ---------------- Output projection: out = y @ w_proj^T (fp32 out) ----------------
__global__ void __launch_bounds__(128)
proj_gemm(const _Float16* __restrict__ yb, const _Float16* __restrict__ wpb,
          float* __restrict__ out) {
  const int lane  = threadIdx.x & 31;
  const int wv    = threadIdx.x >> 5;
  const int mbase = blockIdx.x * 32;
  const int nbase = blockIdx.y * 256 + wv * 64;

  v8f acc[2][4] = {};
  wave_gemm_32x64(yb + (size_t)mbase * N_EMBD,
                  wpb + (size_t)nbase * N_EMBD, lane, acc);

  const int col = lane & 15;
  const int rb  = (lane >> 4) * 8;
#pragma unroll
  for (int i = 0; i < 2; ++i) {
#pragma unroll
    for (int t = 0; t < 4; ++t) {
#pragma unroll
      for (int r = 0; r < 8; ++r) {
        int m = mbase + i * 16 + rb + r;
        int n = nbase + t * 16 + col;
        out[(size_t)m * N_EMBD + n] = acc[i][t][r];
      }
    }
  }
}

extern "C" void kernel_launch(void* const* d_in, const int* in_sizes, int n_in,
                              void* d_out, int out_size, void* d_ws, size_t ws_size,
                              hipStream_t stream) {
  (void)in_sizes; (void)n_in; (void)out_size; (void)ws_size;
  const float* x      = (const float*)d_in[0];
  const float* w_attn = (const float*)d_in[1];
  const float* w_proj = (const float*)d_in[2];
  float* out = (float*)d_out;

  char* ws = (char*)d_ws;
  size_t off = 0;
  auto take = [&](size_t elems) -> _Float16* {
    _Float16* p = (_Float16*)(ws + off);
    off = (off + elems * sizeof(_Float16) + 255) & ~(size_t)255;
    return p;
  };
  const size_t nX  = (size_t)BATCH * SEQ * N_EMBD;        // 4,194,304
  const size_t nWA = (size_t)3 * N_EMBD * N_EMBD;         // 3,145,728
  const size_t nWP = (size_t)N_EMBD * N_EMBD;             // 1,048,576
  const size_t nH  = (size_t)BATCH * N_HEAD * SEQ * HEAD_DIM;  // 4,194,304

  _Float16* xb  = take(nX);
  _Float16* wab = take(nWA);
  _Float16* wpb = take(nWP);
  _Float16* qb  = take(nH);
  _Float16* kb  = take(nH);
  _Float16* vtb = take(nH);
  _Float16* yb  = take(nX);

  // 1) fp32 -> f16 conversions
  cvt_f32_to_f16_x4<<<(int)(nX  / 4 / 256), 256, 0, stream>>>(x,      xb,  (int)(nX  / 4));
  cvt_f32_to_f16_x4<<<(int)(nWA / 4 / 256), 256, 0, stream>>>(w_attn, wab, (int)(nWA / 4));
  cvt_f32_to_f16_x4<<<(int)(nWP / 4 / 256), 256, 0, stream>>>(w_proj, wpb, (int)(nWP / 4));

  // 2) QKV projection: grid = (M/32, N/256), 4 waves/block each doing 32x64
  qkv_gemm<<<dim3(BATCH * SEQ / 32, 3 * N_EMBD / 256), 128, 0, stream>>>(
      xb, wab, qb, kb, vtb);

  // 3) Flash attention: grid = (B*H, T/16/4), 4 query-tiles per block
  attn_flash<<<dim3(BATCH * N_HEAD, SEQ / 16 / 4), 128, 0, stream>>>(
      qb, kb, vtb, yb);

  // 4) Output projection -> fp32 d_out
  proj_gemm<<<dim3(BATCH * SEQ / 32, N_EMBD / 256), 128, 0, stream>>>(
      yb, wpb, out);
}
